// PointNetClassifier_46909632807285
// MI455X (gfx1250) — matrix operand order repaired and checked by reading, not compiled, not run on hardware
//
#include <hip/hip_runtime.h>
#include <hip/hip_bf16.h>
#include <math.h>

#define N_NODES 50000
#define N_EDGES 1600000
#define HIDDEN 64
#define NUM_GRAPHS 64

typedef __attribute__((ext_vector_type(16))) _Float16 v16h;
typedef __attribute__((ext_vector_type(8)))  float    v8f;

union AFrag { v16h v; _Float16 h[16]; };
union CFrag { v8f  v; float     f[8];  };

__global__ void zero_kernel(float* __restrict__ p, int n) {
    int i = blockIdx.x * blockDim.x + threadIdx.x;
    int stride = gridDim.x * blockDim.x;
    for (; i < n; i += stride) p[i] = 0.0f;
}

// Per-edge 2-layer MLP with WMMA, fused relu + scatter-max (as uint atomicMax).
// F_IN: input feature width (3 or 64). KP: K padded to multiple of 32 (32 or 96).
template<int F_IN, int KP>
__global__ __launch_bounds__(256) void edge_mlp_kernel(
    const float* __restrict__ hin,  const float* __restrict__ pos,
    const int*   __restrict__ srcp, const int*   __restrict__ dstp,
    const float* __restrict__ w1,   const float* __restrict__ b1,
    const float* __restrict__ w2,   const float* __restrict__ b2,
    float*       __restrict__ agg)
{
    constexpr int KCH = KP / 32;

    __shared__ _Float16 sW1[KP * 64];
    __shared__ _Float16 sW2[64 * 64];
    __shared__ float    sB1[64];
    __shared__ float    sB2[64];
    __shared__ _Float16 sMid[8][16 * 64];   // per-wave staging of relu(x@W1+b1)

    // Cooperative weight preload (w1 zero-padded past F_IN+3 rows)
    for (int i = threadIdx.x; i < KP * 64; i += 256) {
        int k = i >> 6, n = i & 63;
        sW1[i] = (k < F_IN + 3) ? (_Float16)w1[k * HIDDEN + n] : (_Float16)0.0f;
    }
    for (int i = threadIdx.x; i < 64 * 64; i += 256)
        sW2[i] = (_Float16)w2[i];
    if (threadIdx.x < 64) {
        sB1[threadIdx.x] = b1[threadIdx.x];
        sB2[threadIdx.x] = b2[threadIdx.x];
    }
    __syncthreads();

    const int lane = threadIdx.x & 31;
    const int wave = threadIdx.x >> 5;
    const int hi   = lane >> 4;    // half-wave selector
    const int lr   = lane & 15;    // row (A/C) or column (B) index

    // B fragments, register-resident for the whole edge loop.
    // B k order chosen here (contiguous, same as A): lane element e -> K = hi*16 + e (+32*chunk), N = lr.
    AFrag B1[KCH][4];
#pragma unroll
    for (int c = 0; c < KCH; ++c)
#pragma unroll
        for (int t = 0; t < 4; ++t)
#pragma unroll
            for (int e = 0; e < 16; ++e)
                B1[c][t].h[e] = sW1[(c * 32 + hi * 16 + e) * 64 + t * 16 + lr];

    AFrag B2[2][4];
#pragma unroll
    for (int c = 0; c < 2; ++c)
#pragma unroll
        for (int t = 0; t < 4; ++t)
#pragma unroll
            for (int e = 0; e < 16; ++e)
                B2[c][t].h[e] = sW2[(c * 32 + hi * 16 + e) * 64 + t * 16 + lr];

    _Float16* mid = sMid[wave];

    const int tiles = N_EDGES / 16;   // exactly 100000
    for (int tile = blockIdx.x * 8 + wave; tile < tiles; tile += gridDim.x * 8) {
        const int ebase = tile * 16;
        const int er = ebase + lr;          // this lane's edge (A-matrix row)
        const int s = srcp[er];
        const int d = dstp[er];

        // ---- GEMM1: edge_feat[16xKP] @ W1[KPx64] ----
        CFrag acc[4];
#pragma unroll
        for (int t = 0; t < 4; ++t)
#pragma unroll
            for (int i = 0; i < 8; ++i) acc[t].f[i] = 0.0f;

#pragma unroll
        for (int c = 0; c < KCH; ++c) {
            // A k order chosen here (contiguous, same as B): element e -> K=hi*16+e
            AFrag A;
#pragma unroll
            for (int e = 0; e < 16; ++e) {
                int koff = hi * 16 + e;
                int k = c * 32 + koff;
                float v;
                if (k < F_IN)
                    v = hin[s * F_IN + k];
                else if (k < F_IN + 3)
                    v = pos[s * 3 + (k - F_IN)] - pos[d * 3 + (k - F_IN)];
                else
                    v = 0.0f;
                A.h[e] = (_Float16)v;
            }
#pragma unroll
            for (int t = 0; t < 4; ++t)
                acc[t].v = __builtin_amdgcn_wmma_f32_16x16x32_f16(
                    false, A.v, false, B1[c][t].v, (short)0, acc[t].v, false, false);
        }

        // ---- bias + relu -> f16 staging in LDS (C layout: row = i+8*hi, col = t*16+lr)
#pragma unroll
        for (int t = 0; t < 4; ++t)
#pragma unroll
            for (int i = 0; i < 8; ++i) {
                int row = i + 8 * hi;
                int col = t * 16 + lr;
                float v = acc[t].f[i] + sB1[col];
                mid[row * 64 + col] = (_Float16)(v > 0.0f ? v : 0.0f);
            }
        // same-wave LDS in-order; wait for DS writes before cross-lane reads
        asm volatile("s_wait_dscnt 0" ::: "memory");

        // ---- GEMM2: mid[16x64] @ W2[64x64] ----
        CFrag acc2[4];
#pragma unroll
        for (int t = 0; t < 4; ++t)
#pragma unroll
            for (int i = 0; i < 8; ++i) acc2[t].f[i] = 0.0f;

#pragma unroll
        for (int c = 0; c < 2; ++c) {
            AFrag A2;
#pragma unroll
            for (int e = 0; e < 16; ++e) {
                int koff = hi * 16 + e;
                A2.h[e] = mid[lr * 64 + c * 32 + koff];
            }
#pragma unroll
            for (int t = 0; t < 4; ++t)
                acc2[t].v = __builtin_amdgcn_wmma_f32_16x16x32_f16(
                    false, A2.v, false, B2[c][t].v, (short)0, acc2[t].v, false, false);
        }

        // ---- bias + relu-clamp + scatter-max (uint atomicMax == float max for v>=0)
#pragma unroll
        for (int i = 0; i < 8; ++i) {
            int dn = dstp[ebase + i + 8 * hi];
            unsigned int* drow = (unsigned int*)&agg[dn * HIDDEN];
#pragma unroll
            for (int t = 0; t < 4; ++t) {
                int col = t * 16 + lr;
                float v = acc2[t].f[i] + sB2[col];
                v = (v > 0.0f) ? v : 0.0f;
                atomicMax(drow + col, __float_as_uint(v));
            }
        }
    }
}

// Global max pool per graph (inputs already >= 0)
__global__ void pool_kernel(const float* __restrict__ h, const int* __restrict__ batch,
                            unsigned int* __restrict__ g) {
    int i = blockIdx.x * blockDim.x + threadIdx.x;
    if (i >= N_NODES * HIDDEN) return;
    int node = i >> 6, col = i & 63;
    atomicMax(&g[batch[node] * HIDDEN + col], __float_as_uint(h[i]));
}

// 64x64 @ 64x2 + bias, softmax over 2 classes
__global__ void classify_kernel(const float* __restrict__ g, const float* __restrict__ clw,
                                const float* __restrict__ clb, float* __restrict__ out) {
    int gi = threadIdx.x;
    if (gi >= NUM_GRAPHS) return;
    float a0 = clb[0], a1 = clb[1];
    for (int k = 0; k < HIDDEN; ++k) {
        float v = g[gi * HIDDEN + k];
        a0 += v * clw[k * 2 + 0];
        a1 += v * clw[k * 2 + 1];
    }
    float m = fmaxf(a0, a1);
    float e0 = __expf(a0 - m), e1 = __expf(a1 - m);
    float s = e0 + e1;
    out[gi * 2 + 0] = e0 / s;
    out[gi * 2 + 1] = e1 / s;
}

extern "C" void kernel_launch(void* const* d_in, const int* in_sizes, int n_in,
                              void* d_out, int out_size, void* d_ws, size_t ws_size,
                              hipStream_t stream) {
    const float* x    = (const float*)d_in[0];
    const float* pos  = (const float*)d_in[1];
    const int*   ei   = (const int*)  d_in[2];
    const int*   bat  = (const int*)  d_in[3];
    const float* cw1  = (const float*)d_in[4];
    const float* cb1  = (const float*)d_in[5];
    const float* cw2  = (const float*)d_in[6];
    const float* cb2  = (const float*)d_in[7];
    const float* p0w1 = (const float*)d_in[8];
    const float* p0b1 = (const float*)d_in[9];
    const float* p0w2 = (const float*)d_in[10];
    const float* p0b2 = (const float*)d_in[11];
    const float* p1w1 = (const float*)d_in[12];
    const float* p1b1 = (const float*)d_in[13];
    const float* p1w2 = (const float*)d_in[14];
    const float* p1b2 = (const float*)d_in[15];
    const float* clw  = (const float*)d_in[16];
    const float* clb  = (const float*)d_in[17];

    const int* src = ei;
    const int* dst = ei + N_EDGES;

    float* aggA = (float*)d_ws;
    float* aggB = aggA + (size_t)N_NODES * HIDDEN;
    float* gbuf = aggB + (size_t)N_NODES * HIDDEN;

    dim3 blk(256);
    const int zgrid = 2048;
    const int egrid = 1024;   // 8192 waves, ~12 edge-tiles each

    // layer 1: x(3) -> aggA
    zero_kernel<<<zgrid, blk, 0, stream>>>(aggA, N_NODES * HIDDEN);
    edge_mlp_kernel<3, 32><<<egrid, blk, 0, stream>>>(x, pos, src, dst,
                                                      cw1, cb1, cw2, cb2, aggA);
    // layer 2: aggA(64) -> aggB
    zero_kernel<<<zgrid, blk, 0, stream>>>(aggB, N_NODES * HIDDEN);
    edge_mlp_kernel<64, 96><<<egrid, blk, 0, stream>>>(aggA, pos, src, dst,
                                                       p0w1, p0b1, p0w2, p0b2, aggB);
    // layer 3: aggB(64) -> aggA
    zero_kernel<<<zgrid, blk, 0, stream>>>(aggA, N_NODES * HIDDEN);
    edge_mlp_kernel<64, 96><<<egrid, blk, 0, stream>>>(aggB, pos, src, dst,
                                                       p1w1, p1b1, p1w2, p1b2, aggA);
    // global max pool + classifier
    zero_kernel<<<1, 64, 0, stream>>>(gbuf, NUM_GRAPHS * HIDDEN);
    pool_kernel<<<(N_NODES * HIDDEN + 255) / 256, blk, 0, stream>>>(aggA, bat,
                                                                    (unsigned int*)gbuf);
    classify_kernel<<<1, 64, 0, stream>>>(gbuf, clw, clb, (float*)d_out);
}
